// NBP_oc_33191507264056
// MI455X (gfx1250) — matrix operand order repaired and checked by reading, not compiled, not run on hardware
//
#include <hip/hip_runtime.h>
#include <hip/hip_bf16.h>
#include <math.h>

typedef _Float16 f16;
typedef __attribute__((ext_vector_type(16))) _Float16 v16h;
typedef __attribute__((ext_vector_type(8)))  _Float16 v8h;
typedef __attribute__((ext_vector_type(8)))  float    v8f;

#define NV     2048
#define M1C    1024
#define MOCC   2048
#define BC     16
#define NITERC 8
#define KMAXC  32

__device__ __forceinline__ float softplusf(float x) {
    return (x > 0.f) ? (x + log1pf(expf(-x))) : log1pf(expf(x));
}
__device__ __forceinline__ float fx2f(float a, float b) {
    return fmaxf(a, b) + softplusf(-fabsf(a - b));
}

// ---------------- conversion / setup kernels ----------------

__global__ void k_f32_to_f16(const float* __restrict__ src, f16* __restrict__ dst, int n) {
    int i = blockIdx.x * blockDim.x + threadIdx.x;
    if (i < n) dst[i] = (f16)src[i];
}

__global__ void k_zero(float* __restrict__ p, int n) {
    int i = blockIdx.x * blockDim.x + threadIdx.x;
    if (i < n) p[i] = 0.f;
}

// One thread per check-node row: record nonzero column indices of H=[Hx;Hz].
__global__ void k_rowlist(const float* __restrict__ Hx, const float* __restrict__ Hz,
                          int* __restrict__ row_idx, int* __restrict__ row_cnt) {
    int i = blockIdx.x * blockDim.x + threadIdx.x;
    if (i >= MOCC) return;
    const float* row = (i < M1C) ? (Hx + (size_t)i * NV) : (Hz + (size_t)(i - M1C) * NV);
    int c = 0;
    for (int j = 0; j < NV; ++j) {
        if (row[j] != 0.f && c < KMAXC) row_idx[i * KMAXC + c++] = j;
    }
    row_cnt[i] = c;
}

// One thread per variable-node column: record (row, slot-in-row-list) pairs.
__global__ void k_collist(const float* __restrict__ Hx, const float* __restrict__ Hz,
                          const int* __restrict__ row_idx, const int* __restrict__ row_cnt,
                          int* __restrict__ col_ent, int* __restrict__ col_cnt) {
    int n = blockIdx.x * blockDim.x + threadIdx.x;
    if (n >= NV) return;
    int c = 0;
    for (int i = 0; i < MOCC; ++i) {
        float h = (i < M1C) ? Hx[(size_t)i * NV + n] : Hz[(size_t)(i - M1C) * NV + n];
        if (h != 0.f && c < KMAXC) {
            int slot = 0, cnt = row_cnt[i];
            for (int k = 0; k < cnt; ++k) {
                if (row_idx[i * KMAXC + k] == n) { slot = k; break; }
            }
            col_ent[n * KMAXC + c++] = (i << 6) | slot;
        }
    }
    col_cnt[n] = c;
}

// ---------------- WMMA GEMM: out[16 x 2048] = [A0(16xK)*B0(1024xK)^T | A1*B1^T] ----------------
// One wave (32 lanes) per 16x16 output tile. K looped in steps of 32 with
// v_wmma_f32_16x16x32_f16. Fragment layouts per CDNA5 ISA 7.12.2:
//  A (16x32 f16): lane m=l%15-range holds row m; elems 0..7 = K[k0 + 8*(l/16) .. +7],
//                 elems 8..15 = K[k0 + 16 + 8*(l/16) .. +7]  (two 16B loads)
//  B (32x16 f16): lane holds column n=l%16; elems 0..7 = K[k0 + 8*(l/16) .. +7],
//                 elems 8..15 = K[k0 + 16 + 8*(l/16) .. +7]  (two 16B loads, since B = M^T)
//  C/D (16x16 f32): VGPR r, lane l -> M = r + 8*(l/16), N = l%16
__global__ void k_gemm16(const f16* __restrict__ A0, const f16* __restrict__ B0,
                         const f16* __restrict__ A1, const f16* __restrict__ B1,
                         float* __restrict__ out, int domod) {
    const int tilesPerHalf = M1C / 16;              // 64
    int which = blockIdx.x / tilesPerHalf;          // 0: A0*B0^T, 1: A1*B1^T
    int tile  = blockIdx.x % tilesPerHalf;
    const f16* A  = which ? A1 : A0;
    const f16* Bm = which ? B1 : B0;
    int n0   = tile * 16;
    int lane = threadIdx.x;
    int m    = lane & 15;
    int half = lane >> 4;

    v8f acc = {};
    for (int k0 = 0; k0 < NV; k0 += 32) {
        const v8h a0 = *(const v8h*)(A + (size_t)m * NV + k0 + half * 8);
        const v8h a1 = *(const v8h*)(A + (size_t)m * NV + k0 + 16 + half * 8);
        v16h av;
#pragma unroll
        for (int j = 0; j < 8; ++j) { av[j] = a0[j]; av[j + 8] = a1[j]; }
        const v8h b0 = *(const v8h*)(Bm + (size_t)(n0 + m) * NV + k0 + half * 8);
        const v8h b1 = *(const v8h*)(Bm + (size_t)(n0 + m) * NV + k0 + 16 + half * 8);
        v16h bv;
#pragma unroll
        for (int j = 0; j < 8; ++j) { bv[j] = b0[j]; bv[j + 8] = b1[j]; }
        acc = __builtin_amdgcn_wmma_f32_16x16x32_f16(false, av, false, bv,
                                                     (short)0, acc, false, false);
    }
#pragma unroll
    for (int r = 0; r < 8; ++r) {
        int row = r + half * 8;                     // batch index
        float v = acc[r];
        if (domod) v = fmodf(v, 2.0f);
        out[(size_t)row * MOCC + which * M1C + n0 + m] = v;
    }
}

// ---------------- CN update (per batch x check row) ----------------
__global__ void k_cn(const float* __restrict__ v2c, const int* __restrict__ row_cnt,
                     const float* __restrict__ syn, const float* __restrict__ w_cn,
                     int it, float* __restrict__ c2v) {
    int t = blockIdx.x * blockDim.x + threadIdx.x;
    if (t >= BC * MOCC) return;
    int b = t / MOCC, i = t % MOCC;
    int cnt = row_cnt[i];
    size_t base = ((size_t)b * MOCC + i) * KMAXC;

    float prodsgn = 1.f, A = 0.f;
    for (int k = 0; k < cnt; ++k) {
        float mv = v2c[base + k];
        float sg = (mv == 0.f) ? 1.f : (mv > 0.f ? 1.f : -1.f);
        prodsgn *= sg;
        if (mv != 0.f) A += -logf(tanhf(0.5f * fabsf(mv)));
    }
    float mult = (1.f - 2.f * syn[(size_t)b * MOCC + i]) * w_cn[(size_t)it * MOCC + i];
    for (int k = 0; k < cnt; ++k) {
        float mv = v2c[base + k];
        float sg = (mv == 0.f) ? 1.f : (mv > 0.f ? 1.f : -1.f);
        float phi = (mv != 0.f) ? -logf(tanhf(0.5f * fabsf(mv))) : 0.f;
        float sm = A - phi;
        float second = (sm != 0.f) ? -logf(tanhf(0.5f * sm)) : 0.f;
        c2v[base + k] = prodsgn * sg * second * mult;
    }
}

// ---------------- column sums + VN marginals + Gamma (per batch x variable) ----------------
__global__ void k_colsums(const float* __restrict__ c2v, const int* __restrict__ col_ent,
                          const int* __restrict__ col_cnt, const float* __restrict__ w_llr,
                          int llr_idx, const float* __restrict__ ep_p,
                          float* __restrict__ valup, float* __restrict__ vallo,
                          float* __restrict__ Gamma_out) {
    int t = blockIdx.x * blockDim.x + threadIdx.x;
    if (t >= BC * NV) return;
    int b = t / NV, n = t % NV;
    float s_up = 0.f, s_lo = 0.f;
    int cnt = col_cnt[n];
    for (int k = 0; k < cnt; ++k) {
        int e = col_ent[n * KMAXC + k];
        int i = e >> 6, slot = e & 63;
        float v = c2v[((size_t)b * MOCC + i) * KMAXC + slot];
        if (i < M1C) s_up += v; else s_lo += v;
    }
    float ep  = ep_p[0];
    float llr = logf(3.0f * (1.0f - ep) / ep);
    float base = llr * w_llr[llr_idx];
    float Gz_ = base + s_up;
    float Gx_ = base + s_lo;
    float Gy_ = base + s_up + s_lo;
    valup[t] = softplusf(-Gx_) - fx2f(-Gz_, -Gy_);
    vallo[t] = softplusf(-Gz_) - fx2f(-Gx_, -Gy_);
    Gamma_out[((size_t)b * 3 + 0) * NV + n] = Gy_;
    Gamma_out[((size_t)b * 3 + 1) * NV + n] = Gx_;
    Gamma_out[((size_t)b * 3 + 2) * NV + n] = Gz_;
}

// ---------------- VN update (per batch x check row, gathers weights_vn sparsely) ----------------
__global__ void k_vn(const float* __restrict__ c2v, const int* __restrict__ row_idx,
                     const int* __restrict__ row_cnt, const float* __restrict__ valup,
                     const float* __restrict__ vallo, const float* __restrict__ w_vn,
                     int vn_idx, float* __restrict__ v2c) {
    int t = blockIdx.x * blockDim.x + threadIdx.x;
    if (t >= BC * MOCC) return;
    int b = t / MOCC, i = t % MOCC;
    int cnt = row_cnt[i];
    size_t base = ((size_t)b * MOCC + i) * KMAXC;
    const float* wrow = w_vn + ((size_t)vn_idx * MOCC + i) * NV;
    for (int k = 0; k < cnt; ++k) {
        int j = row_idx[i * KMAXC + k];
        float val = (i < M1C) ? valup[(size_t)b * NV + j] : vallo[(size_t)b * NV + j];
        float mv = val - c2v[base + k];
        mv = fminf(30.f, fmaxf(-30.f, mv));
        v2c[base + k] = mv * wrow[j];
    }
}

// ---------------- loss: build soft-error vectors in f16 for the WMMA GEMM ----------------
__global__ void k_lossprep(const float* __restrict__ Gamma, const float* __restrict__ errx,
                           const float* __restrict__ errz, f16* __restrict__ cz_h,
                           f16* __restrict__ cx_h) {
    int t = blockIdx.x * blockDim.x + threadIdx.x;
    if (t >= BC * NV) return;
    int b = t / NV, n = t % NV;
    float g0 = Gamma[((size_t)b * 3 + 0) * NV + n];   // Gy_
    float g1 = Gamma[((size_t)b * 3 + 1) * NV + n];   // Gx_
    float pX = 1.f / (1.f + expf(g0));                // sigmoid(-g0)
    float pZ = 1.f / (1.f + expf(g1));
    float cz = (errz[(size_t)b * NV + n] == 1.f) ? (1.f - pX) : pX;
    float cx = (errx[(size_t)b * NV + n] == 1.f) ? (1.f - pZ) : pZ;
    cz_h[t] = (f16)cz;
    cx_h[t] = (f16)cx;
}

__global__ void k_lossreduce(const float* __restrict__ synr, float* __restrict__ out, int it) {
    __shared__ float sh[256];
    int b = blockIdx.x;
    float s = 0.f;
    for (int i = threadIdx.x; i < MOCC; i += blockDim.x)
        s += fabsf(sinf(1.57079632679f * synr[(size_t)b * MOCC + i]));
    sh[threadIdx.x] = s;
    __syncthreads();
    for (int w = 128; w > 0; w >>= 1) {
        if ((int)threadIdx.x < w) sh[threadIdx.x] += sh[threadIdx.x + w];
        __syncthreads();
    }
    if (threadIdx.x == 0) out[b * NITERC + it] = sh[0];
}

// ---------------- host-side orchestration ----------------

extern "C" void kernel_launch(void* const* d_in, const int* in_sizes, int n_in,
                              void* d_out, int out_size, void* d_ws, size_t ws_size,
                              hipStream_t stream) {
    const float* errorx = (const float*)d_in[0];
    const float* errorz = (const float*)d_in[1];
    const float* Hx     = (const float*)d_in[2];
    const float* Hz     = (const float*)d_in[3];
    const float* Gx     = (const float*)d_in[4];
    const float* Gz     = (const float*)d_in[5];
    const float* w_vn   = (const float*)d_in[6];
    const float* w_llr  = (const float*)d_in[7];
    const float* w_cn   = (const float*)d_in[8];
    const float* ep     = (const float*)d_in[9];
    (void)in_sizes; (void)n_in; (void)out_size; (void)ws_size;

    float* out   = (float*)d_out;
    float* Gamma = out + BC * NITERC;   // (B,3,N) region of d_out

    char* ws = (char*)d_ws;
    size_t off = 0;
    auto alloc = [&](size_t bytes) -> char* {
        char* p = ws + off;
        off += (bytes + 255) & ~(size_t)255;
        return p;
    };
    f16*   Hx_h   = (f16*)alloc((size_t)M1C * NV * sizeof(f16));
    f16*   Hz_h   = (f16*)alloc((size_t)M1C * NV * sizeof(f16));
    f16*   Gx_h   = (f16*)alloc((size_t)M1C * NV * sizeof(f16));
    f16*   Gz_h   = (f16*)alloc((size_t)M1C * NV * sizeof(f16));
    f16*   ez_h   = (f16*)alloc((size_t)BC * NV * sizeof(f16));
    f16*   ex_h   = (f16*)alloc((size_t)BC * NV * sizeof(f16));
    f16*   cz_h   = (f16*)alloc((size_t)BC * NV * sizeof(f16));
    f16*   cx_h   = (f16*)alloc((size_t)BC * NV * sizeof(f16));
    float* syn    = (float*)alloc((size_t)BC * MOCC * sizeof(float));
    float* synr   = (float*)alloc((size_t)BC * MOCC * sizeof(float));
    int*   rowidx = (int*)alloc((size_t)MOCC * KMAXC * sizeof(int));
    int*   rowcnt = (int*)alloc((size_t)MOCC * sizeof(int));
    int*   colent = (int*)alloc((size_t)NV * KMAXC * sizeof(int));
    int*   colcnt = (int*)alloc((size_t)NV * sizeof(int));
    float* bufA   = (float*)alloc((size_t)BC * MOCC * KMAXC * sizeof(float));  // v2c
    float* bufB   = (float*)alloc((size_t)BC * MOCC * KMAXC * sizeof(float));  // c2v
    float* valup  = (float*)alloc((size_t)BC * NV * sizeof(float));
    float* vallo  = (float*)alloc((size_t)BC * NV * sizeof(float));

    const int TPB = 256;
    const int nMat = M1C * NV;            // 2,097,152
    const int nBN  = BC * NV;             // 32,768
    const int nBM  = BC * MOCC;           // 32,768
    const int nMsg = BC * MOCC * KMAXC;   // 1,048,576

    // 1) f32 -> f16 copies of the matrices / error vectors
    k_f32_to_f16<<<(nMat + TPB - 1) / TPB, TPB, 0, stream>>>(Hx, Hx_h, nMat);
    k_f32_to_f16<<<(nMat + TPB - 1) / TPB, TPB, 0, stream>>>(Hz, Hz_h, nMat);
    k_f32_to_f16<<<(nMat + TPB - 1) / TPB, TPB, 0, stream>>>(Gx, Gx_h, nMat);
    k_f32_to_f16<<<(nMat + TPB - 1) / TPB, TPB, 0, stream>>>(Gz, Gz_h, nMat);
    k_f32_to_f16<<<(nBN + TPB - 1) / TPB, TPB, 0, stream>>>(errorz, ez_h, nBN);
    k_f32_to_f16<<<(nBN + TPB - 1) / TPB, TPB, 0, stream>>>(errorx, ex_h, nBN);

    // 2) sparse structure
    k_rowlist<<<(MOCC + TPB - 1) / TPB, TPB, 0, stream>>>(Hx, Hz, rowidx, rowcnt);
    k_collist<<<(NV + TPB - 1) / TPB, TPB, 0, stream>>>(Hx, Hz, rowidx, rowcnt, colent, colcnt);

    // 3) zero initial c2v messages
    k_zero<<<(nMsg + TPB - 1) / TPB, TPB, 0, stream>>>(bufB, nMsg);

    // 4) syndrome = mod2( [errorz@Hx^T | errorx@Hz^T] ) via WMMA
    k_gemm16<<<2 * (M1C / 16), 32, 0, stream>>>(ez_h, Hx_h, ex_h, Hz_h, syn, 1);

    // 5) initial VN update (msgs0 == 0)
    k_colsums<<<(nBN + TPB - 1) / TPB, TPB, 0, stream>>>(bufB, colent, colcnt, w_llr, 0, ep,
                                                        valup, vallo, Gamma);
    k_vn<<<(nBM + TPB - 1) / TPB, TPB, 0, stream>>>(bufB, rowidx, rowcnt, valup, vallo,
                                                    w_vn, 0, bufA);

    // 6) NITER BP iterations
    for (int it = 0; it < NITERC; ++it) {
        k_cn<<<(nBM + TPB - 1) / TPB, TPB, 0, stream>>>(bufA, rowcnt, syn, w_cn, it, bufB);
        k_colsums<<<(nBN + TPB - 1) / TPB, TPB, 0, stream>>>(bufB, colent, colcnt, w_llr,
                                                            it + 1, ep, valup, vallo, Gamma);
        k_vn<<<(nBM + TPB - 1) / TPB, TPB, 0, stream>>>(bufB, rowidx, rowcnt, valup, vallo,
                                                        w_vn, it + 1, bufA);
        k_lossprep<<<(nBN + TPB - 1) / TPB, TPB, 0, stream>>>(Gamma, errorx, errorz, cz_h, cx_h);
        k_gemm16<<<2 * (M1C / 16), 32, 0, stream>>>(cz_h, Gx_h, cx_h, Gz_h, synr, 0);
        k_lossreduce<<<BC, 256, 0, stream>>>(synr, out, it);
    }
}
